// MitraChannelAttention_73564199846412
// MI455X (gfx1250) — compile-verified, matrix operand order checked by reading, not run on hardware
//
#include <hip/hip_runtime.h>

typedef __attribute__((ext_vector_type(16))) _Float16 v16h;
typedef __attribute__((ext_vector_type(8)))  _Float16 v8h;
typedef __attribute__((ext_vector_type(8)))  float    v8f;
typedef __attribute__((ext_vector_type(4)))  int      v4i;

#define WG_THREADS 128
#define SLOTS_WG   8
#define D_         768
#define EPS        1e-5f

// ---- dynamic LDS layout (bytes) ----
#define OFF_W   0
#define OFF_XF  (128*1024)
#define OFF_A   (160*1024)
#define OFF_U   (176*1024)
#define OFF_P   (240*1024)
#define OFF_B   (244*1024)
#define SMEM_BYTES (244*1024 + 4608)

// ws (f16) layout: [0,65536) qkvo^T (4 x 16384, [n*128+k]); [65536,131072) ff1^T [n*128+k];
//                  [131072,196608) ff2^T [n*512+k]
#define WS_HALVES 196608
#define WS_BYTES  (WS_HALVES * 2)

#if defined(__has_builtin)
#if __has_builtin(__builtin_amdgcn_global_load_async_to_lds_b128) && \
    __has_builtin(__builtin_amdgcn_s_wait_asynccnt)
#define ASYNC_COPY 1
#endif
#endif

#ifdef ASYNC_COPY
typedef __attribute__((address_space(1))) v4i* as1_v4i;
typedef __attribute__((address_space(3))) v4i* as3_v4i;
#endif

__device__ __forceinline__ float wave_sum(float v) {
#pragma unroll
  for (int m = 16; m; m >>= 1) v += __shfl_xor(v, m, 32);
  return v;
}
__device__ __forceinline__ float half_max(float v) {
#pragma unroll
  for (int m = 8; m; m >>= 1) v = fmaxf(v, __shfl_xor(v, m, 16));
  return v;
}
__device__ __forceinline__ float half_sum(float v) {
#pragma unroll
  for (int m = 8; m; m >>= 1) v += __shfl_xor(v, m, 16);
  return v;
}

// 16x32 f16 WMMA operand fragment from LDS (ISA 7.12.2 layout):
// lanes 0-15 hold K 0..7 & 16..23; lanes 16-31 hold K 8..15 & 24..31.
__device__ __forceinline__ v16h lds_frag(const _Float16* p) {
  union { v16h v; v8h h[2]; } u;
  u.h[0] = *(const v8h*)(p);
  u.h[1] = *(const v8h*)(p + 16);
  return u.v;
}

__device__ __forceinline__ v8f wmma32(v16h a, v16h b, v8f c) {
  return __builtin_amdgcn_wmma_f32_16x16x32_f16(false, a, false, b, (short)0, c, false, false);
}

// Copy 65536 halves (8192 x b128) from pre-converted f16 weights into LDS W.
__device__ __forceinline__ void stage_copy(const _Float16* src, _Float16* dst, int tid) {
#ifdef ASYNC_COPY
  for (int i = tid; i < 8192; i += WG_THREADS)
    __builtin_amdgcn_global_load_async_to_lds_b128(
        (as1_v4i)(void*)(src + i * 8), (as3_v4i)(void*)(dst + i * 8), 0, 0);
  __builtin_amdgcn_s_wait_asynccnt(0);
#else
  const v8h* s = (const v8h*)src;
  v8h* d = (v8h*)dst;
  for (int i = tid; i < 8192; i += WG_THREADS) d[i] = s[i];
#endif
}

// ---- pre-pass: convert all weights fp32 -> f16, pre-transposed ----
__global__ void __launch_bounds__(256)
convert_weights_kernel(const float* __restrict__ qw, const float* __restrict__ kw,
                       const float* __restrict__ vw, const float* __restrict__ ow,
                       const float* __restrict__ f1w, const float* __restrict__ f2w,
                       _Float16* __restrict__ ws) {
  int i = blockIdx.x * 256 + threadIdx.x;
  if (i < 65536) {
    int m = i >> 14, r = i & 16383;
    int k = r >> 7, n = r & 127;
    const float* s = (m == 0) ? qw : (m == 1) ? kw : (m == 2) ? vw : ow;
    ws[m * 16384 + n * 128 + k] = (_Float16)s[r];
  } else if (i < 131072) {
    int r = i - 65536;
    int k = r >> 9, n = r & 511;
    ws[65536 + n * 128 + k] = (_Float16)f1w[r];
  } else if (i < WS_HALVES) {
    int r = i - 131072;
    int k = r >> 7, n = r & 127;
    ws[131072 + n * 512 + k] = (_Float16)f2w[r];
  }
}

template <bool PRECONV>
__global__ void __launch_bounds__(WG_THREADS)
mitra_channel_attention_kernel(
    const float* __restrict__ x,
    const float* __restrict__ w1, const float* __restrict__ b1,
    const float* __restrict__ w2, const float* __restrict__ b2,
    const float* __restrict__ qw, const float* __restrict__ qb,
    const float* __restrict__ kw, const float* __restrict__ kb_,
    const float* __restrict__ vw, const float* __restrict__ vb,
    const float* __restrict__ ow, const float* __restrict__ ob,
    const float* __restrict__ f1w, const float* __restrict__ f1b,
    const float* __restrict__ f2w, const float* __restrict__ f2b,
    const _Float16* __restrict__ wsw,
    float* __restrict__ out)
{
  extern __shared__ char smem[];
  _Float16* W  = (_Float16*)(smem + OFF_W);
  float*    XF = (float*)   (smem + OFF_XF);
  _Float16* A_ = (_Float16*)(smem + OFF_A);
  _Float16* U  = (_Float16*)(smem + OFF_U);
  _Float16* P  = (_Float16*)(smem + OFF_P);
  float*    SB = (float*)   (smem + OFF_B);
  _Float16* Q_ = U;               // later reused for attn-output rows
  _Float16* K_ = U + 8192;
  _Float16* VT = U + 16384;       // 16384 halves: [feat 0..127][4 waves * 32 rows]
  _Float16* H1 = U;               // FFN hidden: 64 rows x 512 halves

  const int tid  = threadIdx.x;
  const int lane = tid & 31;
  const int wv   = tid >> 5;
  const int ln16 = lane & 15;
  const int hi8  = ((lane >> 4) & 1) * 8;
  const int rowbase = wv * 16;    // this wave's 16 padded rows (2 slots x 8)
  const int wg = blockIdx.x;

  // ---- step 0: zero K-pad regions; stage biases into LDS
  for (int i = tid; i < 2048;  i += WG_THREADS) P[i]  = (_Float16)0.f;
  for (int i = tid; i < 16384; i += WG_THREADS) VT[i] = (_Float16)0.f;
  for (int i = tid; i < 1152; i += WG_THREADS) {
    float v;
    if (i < 128)       v = qb[i];
    else if (i < 256)  v = kb_[i - 128];
    else if (i < 384)  v = vb[i - 256];
    else if (i < 512)  v = ob[i - 384];
    else if (i < 1024) v = f1b[i - 512];
    else               v = f2b[i - 1024];
    SB[i] = v;
  }

  // ---- stage A weights -> W (qkvo^T, f16 [n*128+k])
  if constexpr (PRECONV) {
    stage_copy(wsw, W, tid);
  } else {
#pragma unroll
    for (int m = 0; m < 4; ++m) {
      const float* s = (m == 0) ? qw : (m == 1) ? kw : (m == 2) ? vw : ow;
      _Float16* dst = W + m * 16384;
      for (int i = tid; i < 16384; i += WG_THREADS) {
        int k = i >> 7, n = i & 127;
        dst[n * 128 + k] = (_Float16)s[i];
      }
    }
  }

  // ---- step 1: LN over D=768 (norm1) then per-row LN (norm2) for this wave's 2 slots
#pragma unroll
  for (int sl = 0; sl < 2; ++sl) {
    const int slot = wg * SLOTS_WG + wv * 2 + sl;
    const long g = (long)slot * D_;
    float xv[24];
    float sum = 0.f, sq = 0.f;
#pragma unroll
    for (int i = 0; i < 24; ++i) {
      float v = x[g + i * 32 + lane];
      xv[i] = v; sum += v; sq += v * v;
    }
    sum = wave_sum(sum); sq = wave_sum(sq);
    float mu = sum * (1.f / 768.f);
    float rs = rsqrtf(sq * (1.f / 768.f) - mu * mu + EPS);
#pragma unroll
    for (int c = 0; c < 8; ++c) {
      int row = rowbase + sl * 8 + c;
      if (c < 6) {
        float y[4]; float s2 = 0.f, q2 = 0.f;
#pragma unroll
        for (int e = 0; e < 4; ++e) {
          int sidx = e * 32 + lane;
          int d = c * 128 + sidx;
          float v = (xv[c * 4 + e] - mu) * rs * w1[d] + b1[d];
          y[e] = v; s2 += v; q2 += v * v;
        }
        s2 = wave_sum(s2); q2 = wave_sum(q2);
        float mu2 = s2 * (1.f / 128.f);
        float rs2 = rsqrtf(q2 * (1.f / 128.f) - mu2 * mu2 + EPS);
#pragma unroll
        for (int e = 0; e < 4; ++e) {
          int sidx = e * 32 + lane;
          float z = (y[e] - mu2) * rs2 * w2[sidx] + b2[sidx];
          XF[row * 128 + sidx] = z;
          A_[row * 128 + sidx] = (_Float16)z;
        }
      } else {
#pragma unroll
        for (int e = 0; e < 4; ++e) {
          int sidx = e * 32 + lane;
          XF[row * 128 + sidx] = 0.f;
          A_[row * 128 + sidx] = (_Float16)0.f;
        }
      }
    }
  }
  __syncthreads();

  // ---- step 3: Q,K,V projections; A-fragments hoisted (invariant over m,t)
  {
    v16h af[4];
    const _Float16* ap = A_ + (rowbase + ln16) * 128 + hi8;
#pragma unroll
    for (int k2 = 0; k2 < 4; ++k2) af[k2] = lds_frag(ap + k2 * 32);
#pragma unroll
    for (int m = 0; m < 3; ++m) {
      const _Float16* wb = W + m * 16384;
      for (int t = 0; t < 8; ++t) {
        v8f acc = {};
        const _Float16* bp = wb + (t * 16 + ln16) * 128 + hi8;
#pragma unroll
        for (int k2 = 0; k2 < 4; ++k2)
          acc = wmma32(af[k2], lds_frag(bp + k2 * 32), acc);
        float bv = SB[m * 128 + t * 16 + ln16];
#pragma unroll
        for (int j = 0; j < 8; ++j) {
          int M = (lane < 16) ? j : j + 8;
          float v = acc[j] + bv;
          if (m == 0)      Q_[(rowbase + M) * 128 + t * 16 + ln16] = (_Float16)v;
          else if (m == 1) K_[(rowbase + M) * 128 + t * 16 + ln16] = (_Float16)v;
          else             VT[(t * 16 + ln16) * 128 + wv * 32 + M] = (_Float16)v;
        }
      }
    }
  }

  // ---- step 4: attention; scores & P@V via WMMA, masked softmax via shuffles
#pragma unroll
  for (int h = 0; h < 2; ++h) {
    const int hb = h * 64;
    v8f sc = {};
    {
      const _Float16* ap = Q_ + (rowbase + ln16) * 128 + hb + hi8;
      const _Float16* bp = K_ + (rowbase + ln16) * 128 + hb + hi8;  // col n = key row n
#pragma unroll
      for (int k2 = 0; k2 < 2; ++k2)
        sc = wmma32(lds_frag(ap + k2 * 32), lds_frag(bp + k2 * 32), sc);
    }
    const int N = ln16;  // key index (D-layout: N across lanes)
#pragma unroll
    for (int j = 0; j < 8; ++j) {
      int M = (lane < 16) ? j : j + 8;  // query row
      bool valid = (((N ^ M) & 8) == 0) && ((N & 7) < 6);  // same slot, real channel
      float val = valid ? sc[j] * 0.125f : -1e30f;          // scale = hd^-0.5
      float mx = half_max(val);
      float p = __expf(val - mx);
      if (!valid) p = 0.f;
      float sm = half_sum(p);
      p = p / sm;
      P[wv * 512 + M * 32 + N] = (_Float16)p;               // K 16..31 stay zero
    }
    {
      v16h pf = lds_frag(P + wv * 512 + ln16 * 32 + hi8);   // P fragment (K pad zeroed)
#pragma unroll
      for (int t = 0; t < 4; ++t) {  // o = P @ V_head
        v8f ov = {};
        const _Float16* bp = VT + (hb + t * 16 + ln16) * 128 + wv * 32 + hi8;
        ov = wmma32(pf, lds_frag(bp), ov);
#pragma unroll
        for (int j = 0; j < 8; ++j) {
          int M = (lane < 16) ? j : j + 8;
          Q_[(rowbase + M) * 128 + hb + t * 16 + ln16] = (_Float16)ov[j]; // o rows -> Q area
        }
      }
    }
  }

  // ---- step 5: output projection + residual into XF (A = o rows, hoisted)
  {
    v16h af[4];
    const _Float16* ap = Q_ + (rowbase + ln16) * 128 + hi8;
#pragma unroll
    for (int k2 = 0; k2 < 4; ++k2) af[k2] = lds_frag(ap + k2 * 32);
    for (int t = 0; t < 8; ++t) {
      v8f acc = {};
      const _Float16* bp = W + 3 * 16384 + (t * 16 + ln16) * 128 + hi8;  // ow^T
#pragma unroll
      for (int k2 = 0; k2 < 4; ++k2)
        acc = wmma32(af[k2], lds_frag(bp + k2 * 32), acc);
      float bv = SB[384 + t * 16 + ln16];
#pragma unroll
      for (int j = 0; j < 8; ++j) {
        int M = (lane < 16) ? j : j + 8;
        XF[(rowbase + M) * 128 + t * 16 + ln16] += acc[j] + bv;
      }
    }
  }

  // ---- step 6: FFN input LN (norm2 params reused) -> A_
  for (int r = 0; r < 16; ++r) {
    int row = rowbase + r;
    if ((r & 7) < 6) {
      float yv[4]; float s2 = 0.f, q2 = 0.f;
#pragma unroll
      for (int e = 0; e < 4; ++e) {
        float v = XF[row * 128 + e * 32 + lane];
        yv[e] = v; s2 += v; q2 += v * v;
      }
      s2 = wave_sum(s2); q2 = wave_sum(q2);
      float mu2 = s2 * (1.f / 128.f);
      float rs2 = rsqrtf(q2 * (1.f / 128.f) - mu2 * mu2 + EPS);
#pragma unroll
      for (int e = 0; e < 4; ++e) {
        int sidx = e * 32 + lane;
        A_[row * 128 + sidx] = (_Float16)((yv[e] - mu2) * rs2 * w2[sidx] + b2[sidx]);
      }
    } else {
#pragma unroll
      for (int e = 0; e < 4; ++e) A_[row * 128 + e * 32 + lane] = (_Float16)0.f;
    }
  }

  __syncthreads();
  // ---- stage B: ff1^T -> W [n*128+k]
  if constexpr (PRECONV) {
    stage_copy(wsw + 65536, W, tid);
  } else {
    for (int i = tid; i < 65536; i += WG_THREADS) {
      int k = i >> 9, n = i & 511;
      W[n * 128 + k] = (_Float16)f1w[i];
    }
  }
  __syncthreads();

  // ---- step 7: FFN1 (K=128 -> N=512) + exact GELU -> H1 (A-fragments hoisted)
  {
    v16h af[4];
    const _Float16* ap = A_ + (rowbase + ln16) * 128 + hi8;
#pragma unroll
    for (int k2 = 0; k2 < 4; ++k2) af[k2] = lds_frag(ap + k2 * 32);
#pragma unroll 2
    for (int t = 0; t < 32; ++t) {
      v8f acc = {};
      const _Float16* bp = W + (t * 16 + ln16) * 128 + hi8;
#pragma unroll
      for (int k2 = 0; k2 < 4; ++k2)
        acc = wmma32(af[k2], lds_frag(bp + k2 * 32), acc);
      float bv = SB[512 + t * 16 + ln16];
#pragma unroll
      for (int j = 0; j < 8; ++j) {
        int M = (lane < 16) ? j : j + 8;
        float v = acc[j] + bv;
        float g = 0.5f * v * (1.f + erff(v * 0.70710678f));
        H1[(rowbase + M) * 512 + t * 16 + ln16] = (_Float16)g;
      }
    }
  }
  __syncthreads();
  // ---- stage C: ff2^T -> W [n*512+k]
  if constexpr (PRECONV) {
    stage_copy(wsw + 131072, W, tid);
  } else {
    for (int i = tid; i < 65536; i += WG_THREADS) {
      int k = i >> 7, n = i & 127;
      W[n * 512 + k] = (_Float16)f2w[i];
    }
  }
  __syncthreads();

  // ---- step 8: FFN2 (K=512 -> N=128) + both residuals + store (A hoisted, 16 frags)
  {
    v16h af[16];
    const _Float16* ap = H1 + (rowbase + ln16) * 512 + hi8;
#pragma unroll
    for (int k2 = 0; k2 < 16; ++k2) af[k2] = lds_frag(ap + k2 * 32);
    for (int t = 0; t < 8; ++t) {
      v8f acc = {};
      const _Float16* bp = W + (t * 16 + ln16) * 512 + hi8;
#pragma unroll
      for (int k2 = 0; k2 < 16; ++k2)
        acc = wmma32(af[k2], lds_frag(bp + k2 * 32), acc);
      float bv = SB[1024 + t * 16 + ln16];
#pragma unroll
      for (int j = 0; j < 8; ++j) {
        int M = (lane < 16) ? j : j + 8;
        int c = M & 7;
        if (c < 6) {
          int slotl = wv * 2 + (M >> 3);
          long gidx = (long)(wg * SLOTS_WG + slotl) * D_ + c * 128 + t * 16 + ln16;
          out[gidx] = x[gidx] + XF[(rowbase + M) * 128 + t * 16 + ln16] + acc[j] + bv;
        }
      }
    }
  }
}

extern "C" void kernel_launch(void* const* d_in, const int* in_sizes, int n_in,
                              void* d_out, int out_size, void* d_ws, size_t ws_size,
                              hipStream_t stream) {
  (void)in_sizes; (void)n_in; (void)out_size;
  const float* x   = (const float*)d_in[0];
  const float* w1  = (const float*)d_in[1];
  const float* b1  = (const float*)d_in[2];
  const float* w2  = (const float*)d_in[3];
  const float* b2  = (const float*)d_in[4];
  const float* qw  = (const float*)d_in[5];
  const float* qb  = (const float*)d_in[6];
  const float* kw  = (const float*)d_in[7];
  const float* kb  = (const float*)d_in[8];
  const float* vw  = (const float*)d_in[9];
  const float* vb  = (const float*)d_in[10];
  const float* ow  = (const float*)d_in[11];
  const float* ob  = (const float*)d_in[12];
  const float* f1w = (const float*)d_in[13];
  const float* f1b = (const float*)d_in[14];
  const float* f2w = (const float*)d_in[15];
  const float* f2b = (const float*)d_in[16];
  float* out = (float*)d_out;

  const int M = 8 * 4096;               // B*N slots
  const int blocks = M / SLOTS_WG;      // 4096 workgroups

  if (ws_size >= (size_t)WS_BYTES) {
    _Float16* ws = (_Float16*)d_ws;
    convert_weights_kernel<<<(WS_HALVES + 255) / 256, 256, 0, stream>>>(
        qw, kw, vw, ow, f1w, f2w, ws);
    (void)hipFuncSetAttribute((const void*)&mitra_channel_attention_kernel<true>,
                              hipFuncAttributeMaxDynamicSharedMemorySize, SMEM_BYTES);
    mitra_channel_attention_kernel<true><<<blocks, WG_THREADS, SMEM_BYTES, stream>>>(
        x, w1, b1, w2, b2, qw, qb, kw, kb, vw, vb, ow, ob,
        f1w, f1b, f2w, f2b, ws, out);
  } else {
    (void)hipFuncSetAttribute((const void*)&mitra_channel_attention_kernel<false>,
                              hipFuncAttributeMaxDynamicSharedMemorySize, SMEM_BYTES);
    mitra_channel_attention_kernel<false><<<blocks, WG_THREADS, SMEM_BYTES, stream>>>(
        x, w1, b1, w2, b2, qw, qb, kw, kb, vw, vb, ow, ob,
        f1w, f1b, f2w, f2b, (const _Float16*)nullptr, out);
  }
}
